// MultiHeadAttention_22316650070222
// MI455X (gfx1250) — compile-verified, matrix operand order checked by reading, not run on hardware
//
#include <hip/hip_runtime.h>
#include <hip/hip_bf16.h>

#define SS 4096
#define EE 2048
#define HH 16
#define DD 128

typedef __attribute__((ext_vector_type(16))) __bf16 v16bf;
typedef __attribute__((ext_vector_type(8)))  float  v8f;

union Bf16x16 {
    v16bf v;
    uint4 u4[2];
};

static __device__ __forceinline__ unsigned short f32_to_bf16(float f) {
    union { float f; unsigned int u; } x;
    x.f = f;
    unsigned int r = x.u + 0x7FFFu + ((x.u >> 16) & 1u);  // round-to-nearest-even
    return (unsigned short)(r >> 16);
}

// Two aligned 16B loads -> one 16x16x32 bf16 fragment (ISA §7.12.2 layout:
// lanes 0-15 hold K=klo..klo+7 and K=klo+16..klo+23 with klo=0, lanes 16-31 with klo=8).
static __device__ __forceinline__ v16bf load_frag(const unsigned short* p) {
    Bf16x16 t;
    t.u4[0] = *(const uint4*)p;
    t.u4[1] = *(const uint4*)(p + 16);
    return t.v;
}

static __device__ __forceinline__ v8f wmma_bf16(v16bf a, v16bf b, v8f c) {
    return __builtin_amdgcn_wmma_f32_16x16x32_bf16(false, a, false, b, (short)0, c,
                                                   false, false);
}

// ---------------------------------------------------------------------------
// fp32 -> bf16 conversion, 4-wide
// ---------------------------------------------------------------------------
__global__ __launch_bounds__(256) void cvt_bf16_kernel(const float4* __restrict__ in,
                                                       unsigned short* __restrict__ out,
                                                       int n4) {
    int i = blockIdx.x * blockDim.x + threadIdx.x;
    if (i < n4) {
        float4 v = in[i];
        union { unsigned short s[4]; uint2 u; } r;
        r.s[0] = f32_to_bf16(v.x);
        r.s[1] = f32_to_bf16(v.y);
        r.s[2] = f32_to_bf16(v.z);
        r.s[3] = f32_to_bf16(v.w);
        *(uint2*)(out + (size_t)i * 4) = r.u;
    }
}

// ---------------------------------------------------------------------------
// QKV projection: qkv = hs @ Wqkv^T + bqkv.
// Each wave computes a 32x64 tile (2 m-tiles x 4 n-tiles), K-loop step 32,
// double-buffered and unrolled x2 so fragment loads overlap the WMMAs.
// Q gets 1/sqrt(D) folded in; V stored transposed per head: vt[j-2E][s].
// ---------------------------------------------------------------------------
__global__ __launch_bounds__(256) void qkv_proj_kernel(const unsigned short* __restrict__ A,
                                                       const unsigned short* __restrict__ W,
                                                       const float* __restrict__ bias,
                                                       unsigned short* __restrict__ qb,
                                                       unsigned short* __restrict__ kb,
                                                       unsigned short* __restrict__ vtb) {
    const int lane = threadIdx.x & 31;
    const int wv   = threadIdx.x >> 5;
    const int ln   = lane & 15;
    const int mh   = (lane >> 4) & 1;
    const int klo  = mh ? 8 : 0;
    const int nstrip = blockIdx.x;                   // 0..95
    const int mpair  = blockIdx.y * 8 + wv;          // 0..127 (pairs of 16-row tiles)

    const unsigned short* pa0 = A + (size_t)(mpair * 32 + ln) * EE + klo;
    const unsigned short* pa1 = A + (size_t)(mpair * 32 + 16 + ln) * EE + klo;
    const unsigned short* pb0 = W + (size_t)(nstrip * 64 + ln) * EE + klo;

    v8f acc[2][4] = {};
    v16bf aA[2], aB[2], bA[4], bB[4];

    // prologue: buffer A holds k-chunk 0
    aA[0] = load_frag(pa0);
    aA[1] = load_frag(pa1);
#pragma unroll
    for (int t = 0; t < 4; ++t) bA[t] = load_frag(pb0 + (size_t)t * 16 * EE);

    for (int k0 = 0; k0 < EE; k0 += 64) {
        // issue loads for k0+32 (buffer B) before consuming buffer A
        aB[0] = load_frag(pa0 + k0 + 32);
        aB[1] = load_frag(pa1 + k0 + 32);
#pragma unroll
        for (int t = 0; t < 4; ++t)
            bB[t] = load_frag(pb0 + (size_t)t * 16 * EE + k0 + 32);
        __builtin_prefetch(pa0 + k0 + 256, 0, 1);
        __builtin_prefetch(pb0 + k0 + 256, 0, 1);
#pragma unroll
        for (int t = 0; t < 4; ++t) {
            acc[0][t] = wmma_bf16(aA[0], bA[t], acc[0][t]);
            acc[1][t] = wmma_bf16(aA[1], bA[t], acc[1][t]);
        }
        if (k0 + 64 < EE) {
            aA[0] = load_frag(pa0 + k0 + 64);
            aA[1] = load_frag(pa1 + k0 + 64);
#pragma unroll
            for (int t = 0; t < 4; ++t)
                bA[t] = load_frag(pb0 + (size_t)t * 16 * EE + k0 + 64);
        }
#pragma unroll
        for (int t = 0; t < 4; ++t) {
            acc[0][t] = wmma_bf16(aB[0], bB[t], acc[0][t]);
            acc[1][t] = wmma_bf16(aB[1], bB[t], acc[1][t]);
        }
    }

    const float inv_sqrt_d = 0.08838834764831845f;  // 1/sqrt(128)
#pragma unroll
    for (int t = 0; t < 4; ++t) {
        const int j = nstrip * 64 + t * 16 + ln;
        const float bj = bias[j];
#pragma unroll
        for (int g = 0; g < 2; ++g) {
#pragma unroll
            for (int r = 0; r < 8; ++r) {
                const int srow = mpair * 32 + g * 16 + r + mh * 8;
                const float v = acc[g][t][r] + bj;
                if (j < EE) {
                    qb[(size_t)srow * EE + j] = f32_to_bf16(v * inv_sqrt_d);
                } else if (j < 2 * EE) {
                    kb[(size_t)srow * EE + (j - EE)] = f32_to_bf16(v);
                } else {
                    vtb[(size_t)(j - 2 * EE) * SS + srow] = f32_to_bf16(v);
                }
            }
        }
    }
}

// ---------------------------------------------------------------------------
// Flash attention: one wave per (16-query tile, head). 32 keys per iteration.
// All 8 K-fragments (and all 8 V-fragments) are batched before the WMMA burst
// so the scheduler can issue partial loadcnt waits; next K/V tiles prefetched.
// ---------------------------------------------------------------------------
__global__ __launch_bounds__(32) void attn_kernel(const unsigned short* __restrict__ q,
                                                  const unsigned short* __restrict__ kmat,
                                                  const unsigned short* __restrict__ vt,
                                                  unsigned short* __restrict__ o) {
    __shared__ __align__(16) unsigned short Pld[16 * 32];

    const int lane = threadIdx.x;
    const int ln   = lane & 15;
    const int mh   = (lane >> 4) & 1;
    const int klo  = mh ? 8 : 0;
    const int qt   = blockIdx.x;    // 0..255
    const int hd   = blockIdx.y;    // 0..15
    const int m0   = qt * 16;

    // Preload Q fragments: 4 K-chunks of 32 over d (Q already scaled by 1/sqrt(D))
    v16bf qf[4];
    {
        const unsigned short* p = q + (size_t)(m0 + ln) * EE + hd * DD + klo;
#pragma unroll
        for (int c = 0; c < 4; ++c) qf[c] = load_frag(p + c * 32);
    }

    v8f acc[8] = {};          // O: 16x128 fp32
    float rowmax[8], rowsum[8];
#pragma unroll
    for (int r = 0; r < 8; ++r) { rowmax[r] = -1e30f; rowsum[r] = 0.0f; }

    const int limit = m0 + 16;                 // keys < limit are causally reachable
    for (int n0 = 0; n0 < limit; n0 += 32) {
        // --- batch-load all 8 K fragments, prefetch next key tile ---
        const unsigned short* pk0 = kmat + (size_t)(n0 + ln) * EE + hd * DD + klo;
        const unsigned short* pk1 = kmat + (size_t)(n0 + 16 + ln) * EE + hd * DD + klo;
        v16bf kf0[4], kf1[4];
#pragma unroll
        for (int c = 0; c < 4; ++c) { kf0[c] = load_frag(pk0 + c * 32); }
#pragma unroll
        for (int c = 0; c < 4; ++c) { kf1[c] = load_frag(pk1 + c * 32); }
        __builtin_prefetch(pk0 + (size_t)32 * EE, 0, 1);
        __builtin_prefetch(pk1 + (size_t)32 * EE, 0, 1);

        v8f s0 = {}, s1 = {};
#pragma unroll
        for (int c = 0; c < 4; ++c) {
            s0 = wmma_bf16(qf[c], kf0[c], s0);
            s1 = wmma_bf16(qf[c], kf1[c], s1);
        }

        // --- causal mask (only near/on the diagonal; wave-uniform branch) ---
        if (n0 + 31 >= m0) {
            const int key0 = n0 + ln;
            const int key1 = key0 + 16;
#pragma unroll
            for (int r = 0; r < 8; ++r) {
                const int row = m0 + r + mh * 8;
                s0[r] = (key0 > row) ? -1e30f : s0[r];
                s1[r] = (key1 > row) ? -1e30f : s1[r];
            }
        }

        // --- online softmax (row reductions within each 16-lane half) ---
        float corr[8];
#pragma unroll
        for (int r = 0; r < 8; ++r) {
            float mx = fmaxf(s0[r], s1[r]);
            mx = fmaxf(mx, __shfl_xor(mx, 1, 32));
            mx = fmaxf(mx, __shfl_xor(mx, 2, 32));
            mx = fmaxf(mx, __shfl_xor(mx, 4, 32));
            mx = fmaxf(mx, __shfl_xor(mx, 8, 32));
            const float nm = fmaxf(rowmax[r], mx);
            corr[r] = __expf(rowmax[r] - nm);
            rowmax[r] = nm;
            s0[r] = __expf(s0[r] - nm);
            s1[r] = __expf(s1[r] - nm);
            float sm = s0[r] + s1[r];
            sm += __shfl_xor(sm, 1, 32);
            sm += __shfl_xor(sm, 2, 32);
            sm += __shfl_xor(sm, 4, 32);
            sm += __shfl_xor(sm, 8, 32);
            rowsum[r] = rowsum[r] * corr[r] + sm;
        }
#pragma unroll
        for (int t = 0; t < 8; ++t)
#pragma unroll
            for (int r = 0; r < 8; ++r) acc[t][r] *= corr[r];

        // --- batch-load all 8 V fragments (V^T layout -> two b128 each) ---
        const unsigned short* pv = vt + (size_t)(hd * DD + ln) * SS + n0 + klo;
        v16bf vf[8];
#pragma unroll
        for (int t = 0; t < 8; ++t) vf[t] = load_frag(pv + (size_t)t * 16 * SS);
        __builtin_prefetch(pv + 32, 0, 1);

        // --- transpose P (C-layout) to A-fragment layout via LDS ---
        __syncthreads();
#pragma unroll
        for (int r = 0; r < 8; ++r) {
            const int m = r + mh * 8;
            Pld[m * 32 + ln]      = f32_to_bf16(s0[r]);
            Pld[m * 32 + 16 + ln] = f32_to_bf16(s1[r]);
        }
        __syncthreads();
        Bf16x16 pa;
        pa.u4[0] = *(const uint4*)(Pld + ln * 32 + klo);
        pa.u4[1] = *(const uint4*)(Pld + ln * 32 + klo + 16);

        // --- O += P @ V ---
#pragma unroll
        for (int t = 0; t < 8; ++t) acc[t] = wmma_bf16(pa.v, vf[t], acc[t]);
    }

    // --- normalize and store (bf16, heads concatenated per token) ---
    float inv[8];
#pragma unroll
    for (int r = 0; r < 8; ++r) inv[r] = 1.0f / rowsum[r];
#pragma unroll
    for (int t = 0; t < 8; ++t) {
        const int dcol = t * 16 + ln;
#pragma unroll
        for (int r = 0; r < 8; ++r) {
            const int srow = m0 + r + mh * 8;
            o[(size_t)srow * EE + hd * DD + dcol] = f32_to_bf16(acc[t][r] * inv[r]);
        }
    }
}

// ---------------------------------------------------------------------------
// Output projection: out = attn @ Wo^T + bo (fp32 store), 32x64 tile per wave,
// double-buffered K-loop identical in structure to qkv_proj_kernel.
// ---------------------------------------------------------------------------
__global__ __launch_bounds__(256) void out_proj_kernel(const unsigned short* __restrict__ A,
                                                       const unsigned short* __restrict__ W,
                                                       const float* __restrict__ bias,
                                                       float* __restrict__ out) {
    const int lane = threadIdx.x & 31;
    const int wv   = threadIdx.x >> 5;
    const int ln   = lane & 15;
    const int mh   = (lane >> 4) & 1;
    const int klo  = mh ? 8 : 0;
    const int nstrip = blockIdx.x;                   // 0..31
    const int mpair  = blockIdx.y * 8 + wv;          // 0..127

    const unsigned short* pa0 = A + (size_t)(mpair * 32 + ln) * EE + klo;
    const unsigned short* pa1 = A + (size_t)(mpair * 32 + 16 + ln) * EE + klo;
    const unsigned short* pb0 = W + (size_t)(nstrip * 64 + ln) * EE + klo;

    v8f acc[2][4] = {};
    v16bf aA[2], aB[2], bA[4], bB[4];

    aA[0] = load_frag(pa0);
    aA[1] = load_frag(pa1);
#pragma unroll
    for (int t = 0; t < 4; ++t) bA[t] = load_frag(pb0 + (size_t)t * 16 * EE);

    for (int k0 = 0; k0 < EE; k0 += 64) {
        aB[0] = load_frag(pa0 + k0 + 32);
        aB[1] = load_frag(pa1 + k0 + 32);
#pragma unroll
        for (int t = 0; t < 4; ++t)
            bB[t] = load_frag(pb0 + (size_t)t * 16 * EE + k0 + 32);
        __builtin_prefetch(pa0 + k0 + 256, 0, 1);
        __builtin_prefetch(pb0 + k0 + 256, 0, 1);
#pragma unroll
        for (int t = 0; t < 4; ++t) {
            acc[0][t] = wmma_bf16(aA[0], bA[t], acc[0][t]);
            acc[1][t] = wmma_bf16(aA[1], bA[t], acc[1][t]);
        }
        if (k0 + 64 < EE) {
            aA[0] = load_frag(pa0 + k0 + 64);
            aA[1] = load_frag(pa1 + k0 + 64);
#pragma unroll
            for (int t = 0; t < 4; ++t)
                bA[t] = load_frag(pb0 + (size_t)t * 16 * EE + k0 + 64);
        }
#pragma unroll
        for (int t = 0; t < 4; ++t) {
            acc[0][t] = wmma_bf16(aB[0], bB[t], acc[0][t]);
            acc[1][t] = wmma_bf16(aB[1], bB[t], acc[1][t]);
        }
    }

#pragma unroll
    for (int t = 0; t < 4; ++t) {
        const int j = nstrip * 64 + t * 16 + ln;
        const float bj = bias[j];
#pragma unroll
        for (int g = 0; g < 2; ++g) {
#pragma unroll
            for (int r = 0; r < 8; ++r) {
                const int srow = mpair * 32 + g * 16 + r + mh * 8;
                out[(size_t)srow * EE + j] = acc[g][t][r] + bj;
            }
        }
    }
}

// ---------------------------------------------------------------------------
extern "C" void kernel_launch(void* const* d_in, const int* in_sizes, int n_in,
                              void* d_out, int out_size, void* d_ws, size_t ws_size,
                              hipStream_t stream) {
    const float* hidden = (const float*)d_in[0];   // [S, E]
    const float* Wqkv   = (const float*)d_in[1];   // [3E, E]
    const float* bqkv   = (const float*)d_in[2];   // [3E]
    const float* Wo     = (const float*)d_in[3];   // [E, E]
    const float* bo     = (const float*)d_in[4];   // [E]
    float* out          = (float*)d_out;           // [S, E]

    char* ws = (char*)d_ws;
    size_t off = 0;
    unsigned short* hs_bf   = (unsigned short*)(ws + off); off += (size_t)SS * EE * 2;
    unsigned short* wqkv_bf = (unsigned short*)(ws + off); off += (size_t)3 * EE * EE * 2;
    unsigned short* wo_bf   = (unsigned short*)(ws + off); off += (size_t)EE * EE * 2;
    unsigned short* q_bf    = (unsigned short*)(ws + off); off += (size_t)SS * EE * 2;
    unsigned short* k_bf    = (unsigned short*)(ws + off); off += (size_t)SS * EE * 2;
    unsigned short* vt_bf   = (unsigned short*)(ws + off); off += (size_t)SS * EE * 2;
    unsigned short* ao_bf   = (unsigned short*)(ws + off); off += (size_t)SS * EE * 2;

    const int n_hs   = SS * EE / 4;
    const int n_wqkv = 3 * EE * EE / 4;
    const int n_wo   = EE * EE / 4;

    cvt_bf16_kernel<<<(n_hs + 255) / 256, 256, 0, stream>>>((const float4*)hidden, hs_bf, n_hs);
    cvt_bf16_kernel<<<(n_wqkv + 255) / 256, 256, 0, stream>>>((const float4*)Wqkv, wqkv_bf, n_wqkv);
    cvt_bf16_kernel<<<(n_wo + 255) / 256, 256, 0, stream>>>((const float4*)Wo, wo_bf, n_wo);

    // QKV GEMM: 96 n-strips of 64 over 6144 cols; 128 m-pairs (32 rows) / 8 waves
    qkv_proj_kernel<<<dim3(96, 16), 256, 0, stream>>>(hs_bf, wqkv_bf, bqkv,
                                                      q_bf, k_bf, vt_bf);

    // Flash attention: (q-tile, head)
    attn_kernel<<<dim3(SS / 16, HH), 32, 0, stream>>>(q_bf, k_bf, vt_bf, ao_bf);

    // Output projection: 2048 cols -> 32 strips
    out_proj_kernel<<<dim3(32, 16), 256, 0, stream>>>(ao_bf, wo_bf, bo, out);
}
